// TopKSAE_32074815767254
// MI455X (gfx1250) — matrix operand order repaired and checked by reading, not compiled, not run on hardware
//
#include <hip/hip_runtime.h>

// ---------------------------------------------------------------------------
// TopK Sparse Autoencoder forward for MI455X (gfx1250, wave32, WMMA).
// Encode GEMM: bf16x3 split (hi/lo) WMMA with f32 accumulate,
// double-buffered LDS + register-staged global loads (2-stage pipeline).
// ---------------------------------------------------------------------------

typedef __bf16          v16bf __attribute__((ext_vector_type(16)));
typedef float           v8f   __attribute__((ext_vector_type(8)));
typedef float           f32x4 __attribute__((ext_vector_type(4)));
typedef unsigned int    u32x4 __attribute__((ext_vector_type(4)));
typedef unsigned short  u16x4 __attribute__((ext_vector_type(4)));

static constexpr int B_   = 8192;
static constexpr int D_   = 768;
static constexpr int H_   = 16384;
static constexpr int KTOP = 64;

static constexpr int BM = 128;   // M tile per workgroup (rows of x)
static constexpr int BN = 128;   // N tile per workgroup (cols = H units)
static constexpr int BK = 32;    // K step (one bf16 WMMA K)
static constexpr int NSTEP = D_ / BK;  // 24
static constexpr int LSTR = 40;  // LDS row stride in ushorts (32 + 8 pad, 16B-aligned)

// ---------------- bf16 helpers (bit-level) ----------------------------------
__device__ __forceinline__ unsigned short f32_to_bf16_rne(float f) {
  unsigned u = __float_as_uint(f);
  unsigned r = u + 0x7FFFu + ((u >> 16) & 1u);
  return (unsigned short)(r >> 16);
}
__device__ __forceinline__ float bf16_bits_to_f32(unsigned short h) {
  return __uint_as_float(((unsigned)h) << 16);
}

union FragU { v16bf v; u32x4 q[2]; };

// A fragment (16x32 bf16, M x K): lane l<16 -> row l, K {0..7, 16..23};
// lane l>=16 -> row l-16, K {8..15, 24..31}.  (ISA 05_wmma 7.12.2)
__device__ __forceinline__ v16bf load_fragA(const unsigned short* rowp, int half) {
  FragU f;
  f.q[0] = *(const u32x4*)(rowp + half * 8);
  f.q[1] = *(const u32x4*)(rowp + 16 + half * 8);
  return f.v;
}
// B fragment (32x16 bf16, K x N), column-major staged rows: lane l<16 ->
// col l, K {0..7, 16..23}; lane l>=16 -> col l-16, K {8..15, 24..31}.
__device__ __forceinline__ v16bf load_fragB(const unsigned short* rowp, int half) {
  FragU f;
  const unsigned short* p = rowp + half * 8;
  f.q[0] = *(const u32x4*)(p);
  f.q[1] = *(const u32x4*)(p + 16);
  return f.v;
}

// ---------------------------------------------------------------------------
// Kernel 1: pre_act = relu(x @ We.T + be)  ->  written to z region of d_out
// ---------------------------------------------------------------------------
__global__ __launch_bounds__(256)
void sae_encode_wmma(const float* __restrict__ x, const float* __restrict__ We,
                     const float* __restrict__ be, float* __restrict__ z) {
  // double-buffered staging: 2 * (4 * 128 * 40 * 2B) = 80 KB
  __shared__ __align__(16) unsigned short Ahi[2][BM][LSTR];
  __shared__ __align__(16) unsigned short Alo[2][BM][LSTR];
  __shared__ __align__(16) unsigned short Bhi[2][BN][LSTR];
  __shared__ __align__(16) unsigned short Blo[2][BN][LSTR];

  const int tid  = threadIdx.x;
  const int lane = tid & 31;
  const int wave = tid >> 5;
  const int wm   = wave & 3;   // 0..3 -> M offset wm*32 (2 subtiles)
  const int wn   = wave >> 2;  // 0..1 -> N offset wn*64 (4 subtiles)
  const int m0   = blockIdx.y * BM;
  const int n0   = blockIdx.x * BN;

  const int half = lane >> 4;
  const int lrow = lane & 15;

  // per-thread staging slots: s = tid + i*256, covering 128x32 f32 as float4s
  int srow[4], scol[4];
#pragma unroll
  for (int i = 0; i < 4; ++i) {
    int s = tid + i * 256;
    srow[i] = s >> 3;
    scol[i] = (s & 7) << 2;
  }

  v8f acc[2][4];
#pragma unroll
  for (int mi = 0; mi < 2; ++mi)
#pragma unroll
    for (int ni = 0; ni < 4; ++ni) {
      v8f zv = {0.f, 0.f, 0.f, 0.f, 0.f, 0.f, 0.f, 0.f};
      acc[mi][ni] = zv;
    }

  // ---- prologue: load K-step 0 into registers ----
  f32x4 va[4], vb[4];
#pragma unroll
  for (int i = 0; i < 4; ++i) {
    va[i] = *(const f32x4*)(x  + (size_t)(m0 + srow[i]) * D_ + scol[i]);
    vb[i] = *(const f32x4*)(We + (size_t)(n0 + srow[i]) * D_ + scol[i]);
  }

  int buf = 0;
  for (int step = 0; step < NSTEP; ++step, buf ^= 1) {
    // ---- convert current registers -> LDS[buf] ----
#pragma unroll
    for (int i = 0; i < 4; ++i) {
      u16x4 ah, al, bh, bl;
#pragma unroll
      for (int m = 0; m < 4; ++m) {
        float fa = va[i][m];
        unsigned short hA = f32_to_bf16_rne(fa);
        ah[m] = hA;
        al[m] = f32_to_bf16_rne(fa - bf16_bits_to_f32(hA));
        float fb = vb[i][m];
        unsigned short hB = f32_to_bf16_rne(fb);
        bh[m] = hB;
        bl[m] = f32_to_bf16_rne(fb - bf16_bits_to_f32(hB));
      }
      *(u16x4*)&Ahi[buf][srow[i]][scol[i]] = ah;
      *(u16x4*)&Alo[buf][srow[i]][scol[i]] = al;
      *(u16x4*)&Bhi[buf][srow[i]][scol[i]] = bh;
      *(u16x4*)&Blo[buf][srow[i]][scol[i]] = bl;
    }
    __syncthreads();  // stores to LDS[buf] visible; also fences buf reuse

    // ---- issue next K-step's global loads (overlap with WMMA below) ----
    if (step + 1 < NSTEP) {
      int kb = (step + 1) * BK;
#pragma unroll
      for (int i = 0; i < 4; ++i) {
        va[i] = *(const f32x4*)(x  + (size_t)(m0 + srow[i]) * D_ + kb + scol[i]);
        vb[i] = *(const f32x4*)(We + (size_t)(n0 + srow[i]) * D_ + kb + scol[i]);
      }
    }

    // ---- load fragments from LDS[buf] ----
    v16bf ah[2], al[2], bh[4], bl[4];
#pragma unroll
    for (int mi = 0; mi < 2; ++mi) {
      int mr = wm * 32 + mi * 16 + lrow;
      ah[mi] = load_fragA(&Ahi[buf][mr][0], half);
      al[mi] = load_fragA(&Alo[buf][mr][0], half);
    }
#pragma unroll
    for (int ni = 0; ni < 4; ++ni) {
      int nr = wn * 64 + ni * 16 + lrow;
      bh[ni] = load_fragB(&Bhi[buf][nr][0], half);
      bl[ni] = load_fragB(&Blo[buf][nr][0], half);
    }

    // ---- 24 WMMAs; consecutive WMMAs hit different accumulators ----
#pragma unroll
    for (int mi = 0; mi < 2; ++mi)
#pragma unroll
      for (int ni = 0; ni < 4; ++ni)
        acc[mi][ni] = __builtin_amdgcn_wmma_f32_16x16x32_bf16(
            false, ah[mi], false, bh[ni], (short)0, acc[mi][ni], false, false);
#pragma unroll
    for (int mi = 0; mi < 2; ++mi)
#pragma unroll
      for (int ni = 0; ni < 4; ++ni)
        acc[mi][ni] = __builtin_amdgcn_wmma_f32_16x16x32_bf16(
            false, ah[mi], false, bl[ni], (short)0, acc[mi][ni], false, false);
#pragma unroll
    for (int mi = 0; mi < 2; ++mi)
#pragma unroll
      for (int ni = 0; ni < 4; ++ni)
        acc[mi][ni] = __builtin_amdgcn_wmma_f32_16x16x32_bf16(
            false, al[mi], false, bh[ni], (short)0, acc[mi][ni], false, false);
  }

  // ---- epilogue: bias + relu, store pre_act into z region ----
  // C layout: lane l, vgpr v -> (M = v + (l>=16)*8, N = l&15)
#pragma unroll
  for (int mi = 0; mi < 2; ++mi) {
#pragma unroll
    for (int ni = 0; ni < 4; ++ni) {
      int n = n0 + wn * 64 + ni * 16 + lrow;
      float bias = be[n];
      int mbase = m0 + wm * 32 + mi * 16 + half * 8;
#pragma unroll
      for (int v = 0; v < 8; ++v) {
        float val = acc[mi][ni][v] + bias;
        z[(size_t)(mbase + v) * H_ + n] = fmaxf(val, 0.0f);
      }
    }
  }
}

// ---------------------------------------------------------------------------
// deterministic exclusive scan over 256 threads (LDS, Hillis-Steele)
// ---------------------------------------------------------------------------
__device__ __forceinline__ int excl_scan256(int v, int tid, int* s) {
  s[tid] = v;
  __syncthreads();
#pragma unroll
  for (int off = 1; off < 256; off <<= 1) {
    int t = (tid >= off) ? s[tid - off] : 0;
    __syncthreads();
    s[tid] += t;
    __syncthreads();
  }
  int incl = s[tid];
  __syncthreads();
  return incl - v;
}

// wave32 sum reduction (5 xor steps)
__device__ __forceinline__ int wave_sum32(int v) {
#pragma unroll
  for (int off = 16; off > 0; off >>= 1) v += __shfl_xor(v, off, 32);
  return v;
}

// ---------------------------------------------------------------------------
// Kernel 2: per-row top-64 mask on z (in place) + sparse decode into x_hat
// ---------------------------------------------------------------------------
__global__ __launch_bounds__(256)
void sae_topk_decode(float* __restrict__ z, const float* __restrict__ Wd,
                     const float* __restrict__ bd, float* __restrict__ xhat) {
  const int b    = blockIdx.x;
  const int tid  = threadIdx.x;
  const int lane = tid & 31;
  float* zrow = z + (size_t)b * H_;

  // row in registers: 16 float4 per thread; element (i,m) -> j = 4*(tid+i*256)+m
  f32x4 r[16];
#pragma unroll
  for (int i = 0; i < 16; ++i)
    r[i] = *(const f32x4*)(zrow + 4 * (tid + i * 256));

  __shared__ int s_cnt;
  __shared__ int s_scan[256];
  __shared__ float s_vals[KTOP];
  __shared__ int   s_idx[KTOP];

  // ReLU output is >= +0, so float order == uint order on the bit patterns.
  // Find T = k-th largest: largest T with count(bits >= T) >= KTOP.
  unsigned lo = 0u, hi = 0x7F800000u;
  while (hi - lo > 1u) {
    unsigned mid = (lo + hi) >> 1;
    if (tid == 0) s_cnt = 0;
    __syncthreads();
    int c = 0;
#pragma unroll
    for (int i = 0; i < 16; ++i)
#pragma unroll
      for (int m = 0; m < 4; ++m)
        c += (__float_as_uint(r[i][m]) >= mid) ? 1 : 0;
    c = wave_sum32(c);
    if (lane == 0) atomicAdd(&s_cnt, c);
    __syncthreads();
    int tot = s_cnt;
    __syncthreads();
    if (tot >= KTOP) lo = mid; else hi = mid;
  }
  // gt = count strictly greater than T (bits >= hi); gt < KTOP by invariant
  if (tid == 0) s_cnt = 0;
  __syncthreads();
  {
    int c = 0;
#pragma unroll
    for (int i = 0; i < 16; ++i)
#pragma unroll
      for (int m = 0; m < 4; ++m)
        c += (__float_as_uint(r[i][m]) >= hi) ? 1 : 0;
    c = wave_sum32(c);
    if (lane == 0) atomicAdd(&s_cnt, c);
  }
  __syncthreads();
  const int gt = s_cnt;
  const int need_eq = KTOP - gt;
  __syncthreads();

  // deterministic tie-break: rank equal-to-T elements by (tid, element) order
  int eq_local = 0;
#pragma unroll
  for (int i = 0; i < 16; ++i)
#pragma unroll
    for (int m = 0; m < 4; ++m)
      eq_local += (__float_as_uint(r[i][m]) == lo) ? 1 : 0;
  const int eq_off = excl_scan256(eq_local, tid, s_scan);

  // pass 1: count local keeps
  int keep_cnt = 0;
  {
    int eqr = 0;
#pragma unroll
    for (int i = 0; i < 16; ++i)
#pragma unroll
      for (int m = 0; m < 4; ++m) {
        unsigned u = __float_as_uint(r[i][m]);
        bool keep = (u >= hi);
        if (!keep && u == lo) { keep = (eq_off + eqr) < need_eq; ++eqr; }
        keep_cnt += keep ? 1 : 0;
      }
  }
  int pos = excl_scan256(keep_cnt, tid, s_scan);

  // pass 2: compact kept (val, idx) into LDS, mask row, write z back
  {
    int eqr = 0;
#pragma unroll
    for (int i = 0; i < 16; ++i) {
#pragma unroll
      for (int m = 0; m < 4; ++m) {
        float v = r[i][m];
        unsigned u = __float_as_uint(v);
        bool keep = (u >= hi);
        if (!keep && u == lo) { keep = (eq_off + eqr) < need_eq; ++eqr; }
        if (keep) {
          if (pos < KTOP) { s_vals[pos] = v; s_idx[pos] = 4 * (tid + i * 256) + m; }
          ++pos;
        }
        r[i][m] = keep ? v : 0.0f;
      }
      *(f32x4*)(zrow + 4 * (tid + i * 256)) = r[i];
    }
  }
  __syncthreads();

  // sparse decode: x_hat[b,d] = bd[d] + sum_i vals[i] * Wd[d, idx[i]]
  for (int d = tid; d < D_; d += 256) {
    float acc = bd[d];
    const float* wrow = Wd + (size_t)d * H_;
#pragma unroll 8
    for (int i = 0; i < KTOP; ++i)
      acc = fmaf(s_vals[i], wrow[s_idx[i]], acc);
    xhat[(size_t)b * D_ + d] = acc;
  }
}

// ---------------------------------------------------------------------------
extern "C" void kernel_launch(void* const* d_in, const int* in_sizes, int n_in,
                              void* d_out, int out_size, void* d_ws, size_t ws_size,
                              hipStream_t stream) {
  (void)in_sizes; (void)n_in; (void)out_size; (void)d_ws; (void)ws_size;
  const float* x  = (const float*)d_in[0];
  const float* We = (const float*)d_in[1];
  const float* be = (const float*)d_in[2];
  const float* Wd = (const float*)d_in[3];
  const float* bd = (const float*)d_in[4];
  // d_in[5] is k (== 64, compiled in as KTOP)

  float* xhat = (float*)d_out;                 // [B, D]
  float* z    = xhat + (size_t)B_ * D_;        // [B, H]

  dim3 gridE(H_ / BN, B_ / BM);                // 128 x 64 workgroups
  sae_encode_wmma<<<gridE, 256, 0, stream>>>(x, We, be, z);
  sae_topk_decode<<<B_, 256, 0, stream>>>(z, Wd, bd, xhat);
}
